// MultiHeadSelfAttention_51110110823124
// MI455X (gfx1250) — matrix operand order repaired and checked by reading, not compiled, not run on hardware
//
#include <hip/hip_runtime.h>
#include <hip/hip_bf16.h>
#include <math.h>

#define DMODEL 1024
#define HEADS  16
#define HDIM   64
#define SEQLEN 2048
#define BATCH  2
#define MROWS  (BATCH * SEQLEN) // 4096

typedef __attribute__((ext_vector_type(8)))  __bf16 v8bf;
typedef __attribute__((ext_vector_type(16))) __bf16 v16bf;
typedef __attribute__((ext_vector_type(8)))  float  v8f;
typedef __attribute__((ext_vector_type(4)))  unsigned uint4v;
typedef __attribute__((ext_vector_type(8)))  int      int8v;
typedef __attribute__((ext_vector_type(4)))  int      int4v;

union F16x16 { v16bf v; v8bf h[2]; };

__device__ inline v8f zero8() {
  v8f z = {0.f, 0.f, 0.f, 0.f, 0.f, 0.f, 0.f, 0.f};
  return z;
}

__device__ inline void wait_ds() {
#if __has_builtin(__builtin_amdgcn_s_wait_dscnt)
  __builtin_amdgcn_s_wait_dscnt(0);
#else
  asm volatile("s_wait_dscnt 0" ::: "memory");
#endif
}

__device__ inline v8f wmma_bf16(v16bf a, v16bf b, v8f c) {
  // D = A(16x32 bf16) * B(32x16 bf16) + C(16x16 f32)
  return __builtin_amdgcn_wmma_f32_16x16x32_bf16(
      /*neg_a=*/false, a, /*neg_b=*/false, b,
      /*c_mod=*/(short)0, c, /*reuse_a=*/false, /*reuse_b=*/false);
}

// Convert 8 consecutive f32 in LDS (16B-aligned) to a v8bf fragment chunk.
__device__ inline v8bf cvt_row8(const float* p) {
  const float4 f0 = *(const float4*)p;
  const float4 f1 = *(const float4*)(p + 4);
  v8bf r;
  r[0] = (__bf16)f0.x; r[1] = (__bf16)f0.y; r[2] = (__bf16)f0.z; r[3] = (__bf16)f0.w;
  r[4] = (__bf16)f1.x; r[5] = (__bf16)f1.y; r[6] = (__bf16)f1.z; r[7] = (__bf16)f1.w;
  return r;
}

// ---------------------------------------------------------------------------
// TDM: DMA a [rows x 32] f32 tile (row stride 1024 elements) from global into
// LDS at ldsOff, padded +4 dwords every 32 dwords -> LDS row stride 36 floats.
// D# built per CDNA5 ISA. Issued once per call (wave op).
// ---------------------------------------------------------------------------
__device__ inline void tdm_load_tile(const float* gsrc, unsigned ldsOff,
                                     int tileRows, int tensorRows) {
  const unsigned long long ga = (unsigned long long)(size_t)gsrc;
  uint4v g0;
  g0[0] = 1u;                                   // count=1 (valid), user mode
  g0[1] = ldsOff;                               // lds_addr (bytes)
  g0[2] = (unsigned)(ga & 0xffffffffu);         // global_addr[31:0]
  g0[3] = (unsigned)(ga >> 32) | (2u << 30);    // global_addr[56:32] | type=2
  int8v g1;
  // data_size=4B(2), pad_enable, pad_interval=code4(32 dw), pad_amount=code3(4 dw)
  g1[0] = (int)((2u << 16) | (1u << 20) | (4u << 22) | (3u << 25));
  g1[1] = (int)((1024u & 0xffffu) << 16);       // tensor_dim0[15:0] @ bits[63:48]
  g1[2] = (int)(((unsigned)tensorRows & 0xffffu) << 16); // tensor_dim1[15:0] @ [95:80]
  g1[3] = (int)(((unsigned)tensorRows >> 16) | (32u << 16)); // dim1 hi | tile_dim0=32
  g1[4] = tileRows;                             // tile_dim1 (tile_dim2=0 unused)
  g1[5] = 1024;                                 // tensor_dim0_stride[31:0]
  g1[6] = 0;                                    // stride hi | tensor_dim1_stride lo
  g1[7] = 0;
  int4v g2 = {0, 0, 0, 0};
  int4v g3 = {0, 0, 0, 0};
#if __has_include(<hip/amd_detail/amd_gfx1250_TDM.h>)
  int8v g4 = {0, 0, 0, 0, 0, 0, 0, 0};
  __builtin_amdgcn_tensor_load_to_lds(g0, g1, g2, g3, g4, 0);
#else
  __builtin_amdgcn_tensor_load_to_lds(g0, g1, g2, g3, 0);
#endif
}

__device__ inline unsigned lds_byte_offset(const void* p) {
  // ISA: flat LDS aperture address has the LDS offset in addr[31:0].
  return (unsigned)(size_t)p;
}

// ---------------------------------------------------------------------------
// GEMM: C[row,col] = sum_k A[row,k] * W[col,k]   (A: [4096,1024] f32, W: [N,1024] f32)
// MODE 0: f32 row-major [4096,1024];  MODE 1: bf16 [b,h,s,d];  MODE 2: bf16 [b,h,d,s]
// Block: 256 threads (8 waves), tile 128(M) x 64(N), K-step 32.
// TDM double-buffered f32 staging; fragments converted f32->bf16 at read.
// ---------------------------------------------------------------------------
template <int MODE>
__global__ __launch_bounds__(256) void gemm_xWT(const float* __restrict__ A,
                                                const float* __restrict__ W,
                                                void* __restrict__ outp) {
  __shared__ __align__(16) float sA[2][128][36]; // padded stride 36 (TDM pad)
  __shared__ __align__(16) float sB[2][64][36];

  const int m0 = blockIdx.x * 128;
  const int n0 = blockIdx.y * 64;
  const int t = threadIdx.x;
  const int wave = t >> 5;
  const int lane = t & 31;
  const int half = lane >> 4;
  const int l15 = lane & 15;

  v8f acc[4];
#pragma unroll
  for (int i = 0; i < 4; ++i) acc[i] = zero8();

  const float* Abase = A + (size_t)m0 * DMODEL;
  const float* Wbase = W + (size_t)n0 * DMODEL;
  unsigned offA[2], offB[2];
  offA[0] = lds_byte_offset(&sA[0][0][0]);
  offA[1] = lds_byte_offset(&sA[1][0][0]);
  offB[0] = lds_byte_offset(&sB[0][0][0]);
  offB[1] = lds_byte_offset(&sB[1][0][0]);

  if (wave == 0) { // prologue: DMA first K-tile into buffer 0
    tdm_load_tile(Abase, offA[0], 128, MROWS);
    tdm_load_tile(Wbase, offB[0], 64, DMODEL);
  }

  constexpr int KSTEPS = DMODEL / 32;
  for (int i = 0; i < KSTEPS; ++i) {
    const int cur = i & 1;
    if (wave == 0) {
      if (i + 1 < KSTEPS) { // prefetch next tile into other buffer, then
        const int k0n = (i + 1) * 32;                   // wait for current:
        tdm_load_tile(Abase + k0n, offA[cur ^ 1], 128, MROWS);
        tdm_load_tile(Wbase + k0n, offB[cur ^ 1], 64, DMODEL);
        __builtin_amdgcn_s_wait_tensorcnt(2); // TDM in-order: first 2 are done
      } else {
        __builtin_amdgcn_s_wait_tensorcnt(0);
      }
    }
    __syncthreads(); // tile `cur` resident for all waves

    // A fragment (16x32): lanes 0-15 K=0..7/16..23, lanes 16-31 K=8..15/24..31
    F16x16 fa;
    const int r = wave * 16 + l15;
    const int ka = half ? 8 : 0;
    fa.h[0] = cvt_row8(&sA[cur][r][ka]);
    fa.h[1] = cvt_row8(&sA[cur][r][16 + ka]);
    const int kb = half ? 8 : 0;
#pragma unroll
    for (int nt = 0; nt < 4; ++nt) {
      F16x16 fb; // B fragment: lane = column, elements = K 8h..8h+7, 16+8h..16+8h+7
      const int n = nt * 16 + l15;
      fb.h[0] = cvt_row8(&sB[cur][n][kb]);
      fb.h[1] = cvt_row8(&sB[cur][n][kb + 16]);
      acc[nt] = wmma_bf16(fa.v, fb.v, acc[nt]);
    }
    __syncthreads(); // all reads of `cur` done before it is re-filled
  }

  // ---- epilogue: C layout (M = rr + 8*half, N = lane&15) ----
#pragma unroll
  for (int nt = 0; nt < 4; ++nt) {
#pragma unroll
    for (int rr = 0; rr < 8; ++rr) {
      const int grow = m0 + wave * 16 + rr + 8 * half;
      const int gcol = n0 + nt * 16 + l15;
      const float val = acc[nt][rr];
      if (MODE == 0) {
        ((float*)outp)[(size_t)grow * DMODEL + gcol] = val;
      } else {
        const int b = grow >> 11, s = grow & (SEQLEN - 1);
        const int h = gcol >> 6, d = gcol & (HDIM - 1);
        if (MODE == 1)
          ((__bf16*)outp)[(((size_t)(b * HEADS + h)) * SEQLEN + s) * HDIM + d] = (__bf16)val;
        else
          ((__bf16*)outp)[(((size_t)(b * HEADS + h)) * HDIM + d) * SEQLEN + s] = (__bf16)val;
      }
    }
  }
}

// ---------------------------------------------------------------------------
// RoPE on bf16 Q and K in [b,h,s,64]; one lane per rotation pair.
// ---------------------------------------------------------------------------
__global__ __launch_bounds__(256) void rope_kernel(__bf16* __restrict__ Q,
                                                   __bf16* __restrict__ K) {
  const int tid = blockIdx.x * blockDim.x + threadIdx.x;
  const int j = tid & 31;   // pair index 0..31
  const int row = tid >> 5; // bh*SEQ + s
  const int s = row & (SEQLEN - 1);
  const float inv = __powf(10000.f, -2.f * (float)j / (float)HDIM);
  const float ang = (float)s * inv;
  float sn, cs;
  __sincosf(ang, &sn, &cs);

  __bf16* q = Q + (size_t)row * HDIM;
  float x1 = (float)q[2 * j], x2 = (float)q[2 * j + 1];
  q[2 * j] = (__bf16)(x1 * cs - x2 * sn);
  q[2 * j + 1] = (__bf16)(x1 * sn + x2 * cs);

  __bf16* k = K + (size_t)row * HDIM;
  x1 = (float)k[2 * j];
  x2 = (float)k[2 * j + 1];
  k[2 * j] = (__bf16)(x1 * cs - x2 * sn);
  k[2 * j + 1] = (__bf16)(x1 * sn + x2 * cs);
}

// ---------------------------------------------------------------------------
// Causal flash attention. Block = 128 threads (4 waves), one (b,h) x 64 query
// rows per block; each wave owns 16 query rows, streams 32-key tiles.
// Q,K: bf16 [b,h,s,64]; Vt: bf16 [b,h,d,s]; O: f32 [b*s, 1024] (h-major cols).
// ---------------------------------------------------------------------------
__global__ __launch_bounds__(128) void attn_fwd(const __bf16* __restrict__ Q,
                                                const __bf16* __restrict__ K,
                                                const __bf16* __restrict__ Vt,
                                                float* __restrict__ O) {
  __shared__ __align__(16) __bf16 sP[4][16][40]; // per-wave P staging (C->A layout)
  const int wave = threadIdx.x >> 5;
  const int lane = threadIdx.x & 31;
  const int half = lane >> 4;
  const int l15 = lane & 15;
  const int bh = blockIdx.y;
  const int b = bh >> 4, h = bh & 15;
  const int qBase = blockIdx.x * 64 + wave * 16;

  const __bf16* Qp = Q + (size_t)bh * SEQLEN * HDIM;
  const __bf16* Kp = K + (size_t)bh * SEQLEN * HDIM;
  const __bf16* Vp = Vt + (size_t)bh * HDIM * SEQLEN;

  const int ka = half ? 8 : 0;
  F16x16 aq[2];
#pragma unroll
  for (int c = 0; c < 2; ++c) {
    const __bf16* p = Qp + (size_t)(qBase + l15) * HDIM + 32 * c;
    aq[c].h[0] = *(const v8bf*)(p + ka);
    aq[c].h[1] = *(const v8bf*)(p + 16 + ka);
  }

  float mrow[8], lrow[8];
  v8f o[4];
#pragma unroll
  for (int i = 0; i < 8; ++i) { mrow[i] = -1e30f; lrow[i] = 0.f; }
#pragma unroll
  for (int i = 0; i < 4; ++i) o[i] = zero8();

  const float scl = 0.125f; // 1/sqrt(64)

  for (int kb = 0; kb < qBase + 16; kb += 32) {
    // ---- S = Q * K^T for 32 keys: 2 output tiles x 2 contract chunks ----
    v8f sc[2];
#pragma unroll
    for (int nt = 0; nt < 2; ++nt) {
      sc[nt] = zero8();
#pragma unroll
      for (int c = 0; c < 2; ++c) {
        F16x16 fb; // B-frag: lane = key column, elements = d 8h..8h+7, 16+8h..16+8h+7
        const __bf16* p =
            Kp + (size_t)(kb + nt * 16 + l15) * HDIM + 32 * c + (half ? 8 : 0);
        fb.h[0] = *(const v8bf*)p;
        fb.h[1] = *(const v8bf*)(p + 16);
        sc[nt] = wmma_bf16(aq[c].v, fb.v, sc[nt]);
      }
    }
    // ---- scale + causal mask + online softmax (row across 16 lanes) ----
#pragma unroll
    for (int r = 0; r < 8; ++r) {
      const int row = qBase + 8 * half + r;
      float v0 = sc[0][r] * scl;
      if (kb + l15 > row) v0 = -1e30f;
      float v1 = sc[1][r] * scl;
      if (kb + 16 + l15 > row) v1 = -1e30f;
      float rm = fmaxf(v0, v1);
#pragma unroll
      for (int off = 1; off < 16; off <<= 1)
        rm = fmaxf(rm, __shfl_xor(rm, off, 32));
      const float nm = fmaxf(mrow[r], rm);
      const float corr = __expf(mrow[r] - nm);
      mrow[r] = nm;
      const float p0 = __expf(v0 - nm);
      const float p1 = __expf(v1 - nm);
      float ps = p0 + p1;
#pragma unroll
      for (int off = 1; off < 16; off <<= 1) ps += __shfl_xor(ps, off, 32);
      lrow[r] = lrow[r] * corr + ps;
#pragma unroll
      for (int nt = 0; nt < 4; ++nt) o[nt][r] *= corr;
      sP[wave][8 * half + r][l15] = (__bf16)p0;
      sP[wave][8 * half + r][l15 + 16] = (__bf16)p1;
    }
    wait_ds(); // per-wave LDS RAW fence (DS ops are in-order within a wave)

    // ---- O += P * V : P re-read in A-fragment layout, V frags from [d,s] ----
    F16x16 ap;
    ap.h[0] = *(const v8bf*)&sP[wave][l15][ka];
    ap.h[1] = *(const v8bf*)&sP[wave][l15][16 + ka];
    const int kv = kb + (half ? 8 : 0);
#pragma unroll
    for (int nt = 0; nt < 4; ++nt) {
      F16x16 fv; // B-frag: lane = d column, elements = keys 8h..8h+7, 16+8h..16+8h+7
      const __bf16* p = Vp + (size_t)(nt * 16 + l15) * SEQLEN + kv;
      fv.h[0] = *(const v8bf*)p;
      fv.h[1] = *(const v8bf*)(p + 16);
      o[nt] = wmma_bf16(ap.v, fv.v, o[nt]);
    }
  }

  // ---- normalize and store f32 [b*s, h*64+d] ----
#pragma unroll
  for (int r = 0; r < 8; ++r) {
    const float inv = 1.0f / lrow[r];
    const int srow = qBase + 8 * half + r;
#pragma unroll
    for (int nt = 0; nt < 4; ++nt) {
      O[((size_t)(b * SEQLEN + srow)) * DMODEL + h * HDIM + nt * 16 + l15] =
          o[nt][r] * inv;
    }
  }
}

// ---------------------------------------------------------------------------
extern "C" void kernel_launch(void* const* d_in, const int* in_sizes, int n_in,
                              void* d_out, int out_size, void* d_ws, size_t ws_size,
                              hipStream_t stream) {
  (void)in_sizes; (void)n_in; (void)out_size; (void)ws_size;
  const float* x  = (const float*)d_in[0];
  const float* Wq = (const float*)d_in[1];
  const float* Wk = (const float*)d_in[2];
  const float* Wv = (const float*)d_in[3];
  const float* Wo = (const float*)d_in[4];
  // d_in[5] = token_positions (arange) — positions derived from s index directly.

  char* w = (char*)d_ws;
  const size_t QKV_BYTES = (size_t)BATCH * HEADS * SEQLEN * HDIM * sizeof(__bf16); // 8 MiB
  __bf16* Qb = (__bf16*)(w);
  __bf16* Kb = (__bf16*)(w + QKV_BYTES);
  __bf16* Vt = (__bf16*)(w + 2 * QKV_BYTES);
  float*  At = (float*)(w + 3 * QKV_BYTES); // 16 MiB attention output, f32

  dim3 gg(MROWS / 128, DMODEL / 64);
  gemm_xWT<1><<<gg, 256, 0, stream>>>(x, Wq, Qb);
  gemm_xWT<1><<<gg, 256, 0, stream>>>(x, Wk, Kb);
  gemm_xWT<2><<<gg, 256, 0, stream>>>(x, Wv, Vt);

  const int rope_threads = BATCH * HEADS * SEQLEN * 32;
  rope_kernel<<<rope_threads / 256, 256, 0, stream>>>(Qb, Kb);

  attn_fwd<<<dim3(SEQLEN / 64, BATCH * HEADS), 128, 0, stream>>>(Qb, Kb, Vt, At);

  gemm_xWT<0><<<gg, 256, 0, stream>>>(At, Wo, (float*)d_out);
}
